// AttentionModel_85950885527881
// MI455X (gfx1250) — matrix operand order repaired and checked by reading, not compiled, not run on hardware
//
#include <hip/hip_runtime.h>
#include <hip/hip_bf16.h>
#include <cstdint>
#include <cstddef>

// ---------------------------------------------------------------------------
// Types matching CDNA5 WMMA fragment shapes (wave32):
//   A: 16x32 f16 -> 16 halfs / lane (8 VGPRs)
//   B: 32x16 f16 -> 16 halfs / lane (8 VGPRs)
//   C/D f32: 16x16 -> 8 floats / lane (8 VGPRs)
//   C/D f16: 16x16 -> 8 halfs  / lane (4 VGPRs)
// ---------------------------------------------------------------------------
typedef _Float16 h2  __attribute__((ext_vector_type(2)));
typedef _Float16 h8  __attribute__((ext_vector_type(8)));
typedef _Float16 h16 __attribute__((ext_vector_type(16)));
typedef float    f8  __attribute__((ext_vector_type(8)));
typedef unsigned int u32x4 __attribute__((ext_vector_type(4)));
typedef int      i32x4 __attribute__((ext_vector_type(4)));
typedef int      i32x8 __attribute__((ext_vector_type(8)));

#define DH 512   // D == H == 512
#define NSEQ 8192
#define BN 32    // keys per attention iteration

#if __has_builtin(__builtin_amdgcn_tensor_load_to_lds) && \
    __has_builtin(__builtin_amdgcn_s_wait_tensorcnt)
#define HAVE_TDM 1
#endif
#if __has_builtin(__builtin_amdgcn_wmma_f16_16x16x32_f16)
#define HAVE_WMMA_F16C 1
#endif

__device__ __forceinline__ h16 cat16(h8 lo, h8 hi) {
  return __builtin_shufflevector(lo, hi, 0,1,2,3,4,5,6,7,8,9,10,11,12,13,14,15);
}

// A-fragment: per lane, K = kh*16 + 0..15; caller folds kh*16.
__device__ __forceinline__ h16 load_a(const _Float16* p) {
  h8 lo = *(const h8*)p;
  h8 hi = *(const h8*)(p + 8);
  return cat16(lo, hi);
}

// B-fragment: per lane (column n), K = kh*16 + 0..15; caller folds kh*16.
__device__ __forceinline__ h16 load_b(const _Float16* p) {
  h8 lo = *(const h8*)p;
  h8 hi = *(const h8*)(p + 8);
  return cat16(lo, hi);
}

__device__ __forceinline__ f8 fzero8() {
  f8 z;
#pragma unroll
  for (int j = 0; j < 8; ++j) z[j] = 0.0f;
  return z;
}

__device__ __forceinline__ f8 wmma_f32c(h16 a, h16 b, f8 c) {
  return __builtin_amdgcn_wmma_f32_16x16x32_f16(false, a, false, b, (short)0, c,
                                                false, false);
}

// ---------------------------------------------------------------------------
// fp32 -> f16 elementwise convert
// ---------------------------------------------------------------------------
__global__ void k_convert_f16(const float* __restrict__ src,
                              _Float16* __restrict__ dst, int n) {
  int i = blockIdx.x * blockDim.x + threadIdx.x;
  if (i < n) dst[i] = (_Float16)src[i];
}

// ---------------------------------------------------------------------------
// W[512][512] f32 -> WT[512][512] f16   (WT[n][k] = W[k][n])
// ---------------------------------------------------------------------------
__global__ void k_transpose_f16(const float* __restrict__ W,
                                _Float16* __restrict__ WT) {
  int n = blockIdx.x * 16 + threadIdx.x;
  int k = blockIdx.y * 16 + threadIdx.y;
  WT[(size_t)n * DH + k] = (_Float16)W[(size_t)k * DH + n];
}

// ---------------------------------------------------------------------------
// GEMM: out = (A[8192x512]f16 @ BT^T + bias) * scale
// Wave tile 16 rows x 128 cols (8 WMMAs per A-fragment); all 8 B-fragments
// loaded before the WMMA burst so one s_wait covers the whole chunk.
// Block 256 thr = 8 waves -> 128x128 tile; grid (64, 4).
// ---------------------------------------------------------------------------
__global__ void __launch_bounds__(256)
k_gemm_wmma(const _Float16* __restrict__ A, const _Float16* __restrict__ BT,
            const float* __restrict__ bias, float scale,
            _Float16* __restrict__ outH, float* __restrict__ outF, int Kdim) {
  const int lane = threadIdx.x & 31;
  const int wave = threadIdx.x >> 5;
  const int ln   = lane & 15;
  const int kh   = lane >> 4;
  const int row0 = blockIdx.x * 128 + wave * 16;
  const int col0 = blockIdx.y * 128;

  f8 acc[8];
#pragma unroll
  for (int t = 0; t < 8; ++t) acc[t] = fzero8();

  const _Float16* arow  = A + (size_t)(row0 + ln) * Kdim + kh * 16;
  const _Float16* bbase = BT + (size_t)(col0 + ln) * Kdim + kh * 16;

  for (int kb = 0; kb < Kdim; kb += 32) {
    h16 a = load_a(arow + kb);
    h16 b[8];
#pragma unroll
    for (int t = 0; t < 8; ++t)
      b[t] = load_b(bbase + (size_t)(t * 16) * Kdim + kb);
#pragma unroll
    for (int t = 0; t < 8; ++t)
      acc[t] = wmma_f32c(a, b[t], acc[t]);
  }

#pragma unroll
  for (int t = 0; t < 8; ++t) {
    int col = col0 + t * 16 + ln;
    float bv = bias[col];
#pragma unroll
    for (int j = 0; j < 8; ++j) {
      int row = row0 + j + 8 * kh;
      float r = (acc[t][j] + bv) * scale;
      if (outH) outH[(size_t)row * DH + col] = (_Float16)r;
      else      outF[(size_t)row * DH + col] = r;
    }
  }
}

// ---------------------------------------------------------------------------
// Flash attention: attended = softmax(q @ k^T) @ v   (1/sqrt(H) folded into q)
// Block: 256 thr = 8 waves, 128 query rows/block (16 per wave), grid 64.
// Per 32-key iteration:
//   - K block (32x512 f16, 32 KB) DMA'd into LDS by the Tensor Data Mover
//     (one tensor_load_to_lds by wave 0, s_wait_tensorcnt), fallback copy.
//   - V block staged transposed into LDS with key-pair packed b32 stores.
//   - Online softmax per wave (rows wave-private, lane-shuffle reductions).
//   - P bounced through per-wave LDS (C-layout -> A-layout).
//   - O accumulated with V_WMMA_F16_16X16X32_F16 when available (alpha-rescale
//     fused into the f16 C operand), else f32-WMMA + f16 storage.
// LDS: sK 32K + sVT 32K + sP 8K = 73728 bytes (dynamic).
// ---------------------------------------------------------------------------
__global__ void __launch_bounds__(256)
k_attention(const _Float16* __restrict__ q16, const _Float16* __restrict__ k16,
            const _Float16* __restrict__ v16, _Float16* __restrict__ out16) {
  extern __shared__ _Float16 smem[];
  _Float16* sK  = smem;                 // [32][512]   K block, row-major
  _Float16* sVT = smem + BN * DH;       // [512][32]   V block, transposed
  const int tid  = threadIdx.x;
  const int lane = tid & 31;
  const int wave = tid >> 5;
  const int ln   = lane & 15;
  const int kh   = lane >> 4;
  _Float16* sPw = smem + 2 * BN * DH + wave * (16 * BN);

  const int qrow0 = blockIdx.x * 128 + wave * 16;
  const _Float16* qrow = q16 + (size_t)(qrow0 + ln) * DH + kh * 16;

  // Output accumulator, f16 storage; element j <-> row j + 8*kh (matches both
  // the f16 C/D layout and the f32 C-layout slot convention).
  h8 Oh[32];
#pragma unroll
  for (int t = 0; t < 32; ++t)
#pragma unroll
    for (int e = 0; e < 8; ++e) Oh[t][e] = (_Float16)0.0f;

  float mstate[8], lstate[8];
#pragma unroll
  for (int j = 0; j < 8; ++j) { mstate[j] = -1.0e30f; lstate[j] = 0.0f; }

  for (int kb = 0; kb < NSEQ; kb += BN) {
    __syncthreads();  // previous iteration's LDS readers are done

    // ---- stage V block transposed: sVT[h][key] = v16[kb+key][h] ----
    {
      int kp = tid & 15;            // key pair: keys 2kp, 2kp+1
      int hs = (tid >> 4) * 32;     // 32-h segment
      const h8* g0 = (const h8*)(v16 + (size_t)(kb + 2 * kp) * DH + hs);
      const h8* g1 = (const h8*)(v16 + (size_t)(kb + 2 * kp + 1) * DH + hs);
#pragma unroll
      for (int i = 0; i < 4; ++i) {
        h8 r0 = g0[i], r1 = g1[i];
#pragma unroll
        for (int e = 0; e < 8; ++e) {
          h2 pr; pr[0] = r0[e]; pr[1] = r1[e];
          *(h2*)(sVT + (hs + i * 8 + e) * BN + 2 * kp) = pr;
        }
      }
    }

    // ---- stage K block via Tensor Data Mover (wave 0 only) ----
#if defined(HAVE_TDM)
    if (wave == 0) {
      unsigned long long ga =
          (unsigned long long)(const void*)(k16 + (size_t)kb * DH);
      u32x4 g0d;
      g0d[0] = 1u;                                   // count=1 (user D#)
      g0d[1] = (unsigned)(size_t)(void*)sK;          // lds_addr (bytes)
      g0d[2] = (unsigned)(ga & 0xFFFFFFFFu);         // global_addr[31:0]
      g0d[3] = (unsigned)((ga >> 32) & 0x01FFFFFFu)  // global_addr[56:32]
               | 0x80000000u;                        // type=2 (image)
      i32x8 g1d;
      g1d[0] = 1 << 16;              // data_size=1 -> 2-byte elements
      g1d[1] = (int)(512u << 16);    // tensor_dim0 = 512 (bits 79:48, low part)
      g1d[2] = (int)(8192u << 16);   // tensor_dim1 = 8192 (bits 111:80, low)
      g1d[3] = (int)(512u << 16);    // tile_dim0 = 512 (bits 127:112)
      g1d[4] = BN;                   // tile_dim1 = 32 (bits 143:128)
      g1d[5] = 512;                  // tensor_dim0_stride = 512 (bits 207:160)
      g1d[6] = 0;
      g1d[7] = 0;
      i32x4 z4; z4[0] = 0; z4[1] = 0; z4[2] = 0; z4[3] = 0;
      i32x8 z8;
#pragma unroll
      for (int i = 0; i < 8; ++i) z8[i] = 0;
      // 6-arg form on this toolchain: (g0, g1, g2, g3, extra, cpol)
      __builtin_amdgcn_tensor_load_to_lds(g0d, g1d, z4, z4, z8, 0);
      __builtin_amdgcn_s_wait_tensorcnt(0);
    }
#else
    {
      int key = tid >> 3;
      int hsk = (tid & 7) * 64;
      const h8* g = (const h8*)(k16 + (size_t)(kb + key) * DH + hsk);
      h8* s = (h8*)(sK + key * DH + hsk);
#pragma unroll
      for (int i = 0; i < 8; ++i) s[i] = g[i];
    }
#endif
    __syncthreads();

    // ---- scores S = q @ k^T : 16 rows x 32 keys (two 16x16 tiles) ----
    f8 s0 = fzero8(), s1 = fzero8();
#pragma unroll
    for (int hb = 0; hb < DH; hb += 32) {
      h16 a  = load_a(qrow + hb);
      h16 b0 = load_b(sK + ln * DH + hb + kh * 16);
      h16 b1 = load_b(sK + (16 + ln) * DH + hb + kh * 16);
      s0 = wmma_f32c(a, b0, s0);
      s1 = wmma_f32c(a, b1, s1);
    }

    // ---- online softmax (row j+8*kh lives in slot j across 16 lanes) ----
    float rm[8], al[8], rs[8];
#pragma unroll
    for (int j = 0; j < 8; ++j) rm[j] = fmaxf(s0[j], s1[j]);
#pragma unroll
    for (int msk = 1; msk <= 8; msk <<= 1)
#pragma unroll
      for (int j = 0; j < 8; ++j) rm[j] = fmaxf(rm[j], __shfl_xor(rm[j], msk, 32));

#pragma unroll
    for (int j = 0; j < 8; ++j) {
      float mn = fmaxf(mstate[j], rm[j]);
      al[j] = __expf(mstate[j] - mn);
      mstate[j] = mn;
      float p0 = __expf(s0[j] - mn);
      float p1 = __expf(s1[j] - mn);
      rs[j] = p0 + p1;
      sPw[(j + 8 * kh) * BN + ln]      = (_Float16)p0;
      sPw[(j + 8 * kh) * BN + 16 + ln] = (_Float16)p1;
    }
#pragma unroll
    for (int msk = 1; msk <= 8; msk <<= 1)
#pragma unroll
      for (int j = 0; j < 8; ++j) rs[j] += __shfl_xor(rs[j], msk, 32);
#pragma unroll
    for (int j = 0; j < 8; ++j) lstate[j] = al[j] * lstate[j] + rs[j];

    __syncthreads();  // order P writes (C-layout) before A-layout reads

    // ---- O = P @ V + alpha * O ----
    h16 pa = load_a(sPw + ln * BN + kh * 16);
#if defined(HAVE_WMMA_F16C)
    _Float16 alh[8];
#pragma unroll
    for (int j = 0; j < 8; ++j) alh[j] = (_Float16)al[j];
#pragma unroll
    for (int t = 0; t < 32; ++t) {
      h16 bv = load_b(sVT + (t * 16 + ln) * BN + kh * 16);
      h8 c = Oh[t];
#pragma unroll
      for (int j = 0; j < 8; ++j) c[j] = c[j] * alh[j];
      Oh[t] = __builtin_amdgcn_wmma_f16_16x16x32_f16(false, pa, false, bv,
                                                     (short)0, c, false, false);
    }
#else
#pragma unroll
    for (int t = 0; t < 32; ++t) {
      h16 bv = load_b(sVT + (t * 16 + ln) * BN + kh * 16);
      f8 c;
#pragma unroll
      for (int j = 0; j < 8; ++j) c[j] = al[j] * (float)Oh[t][j];
      c = wmma_f32c(pa, bv, c);
#pragma unroll
      for (int j = 0; j < 8; ++j) Oh[t][j] = (_Float16)c[j];
    }
#endif
  }

  // ---- normalize and store attended (f16) ----
  float linv[8];
#pragma unroll
  for (int j = 0; j < 8; ++j) linv[j] = 1.0f / lstate[j];
#pragma unroll
  for (int t = 0; t < 32; ++t)
#pragma unroll
    for (int j = 0; j < 8; ++j) {
      size_t idx = (size_t)(qrow0 + j + 8 * kh) * DH + t * 16 + ln;
      out16[idx] = (_Float16)((float)Oh[t][j] * linv[j]);
    }
}

// ---------------------------------------------------------------------------
// Host launcher
// ---------------------------------------------------------------------------
extern "C" void kernel_launch(void* const* d_in, const int* in_sizes, int n_in,
                              void* d_out, int out_size, void* d_ws, size_t ws_size,
                              hipStream_t stream) {
  (void)in_sizes; (void)n_in; (void)out_size; (void)ws_size;
  const float* x  = (const float*)d_in[0];
  const float* Wq = (const float*)d_in[1];
  const float* bq = (const float*)d_in[2];
  const float* Wk = (const float*)d_in[3];
  const float* bk = (const float*)d_in[4];
  const float* Wv = (const float*)d_in[5];
  const float* bv = (const float*)d_in[6];
  const float* Wo = (const float*)d_in[7];
  const float* bo = (const float*)d_in[8];
  float* out = (float*)d_out;

  const size_t ND = (size_t)NSEQ * DH;   // 4M elements
  _Float16* ws  = (_Float16*)d_ws;
  _Float16* x16 = ws;            // [8192,512]
  _Float16* q16 = x16 + ND;
  _Float16* k16 = q16 + ND;
  _Float16* v16 = k16 + ND;
  _Float16* a16 = v16 + ND;      // attended
  _Float16* WqT = a16 + ND;      // [512,512] each, transposed
  _Float16* WkT = WqT + (size_t)DH * DH;
  _Float16* WvT = WkT + (size_t)DH * DH;
  _Float16* WoT = WvT + (size_t)DH * DH;

  // 1) precision convert + weight transpose
  k_convert_f16<<<(int)((ND + 255) / 256), 256, 0, stream>>>(x, x16, (int)ND);
  dim3 tb(16, 16), tg(32, 32);
  k_transpose_f16<<<tg, tb, 0, stream>>>(Wq, WqT);
  k_transpose_f16<<<tg, tb, 0, stream>>>(Wk, WkT);
  k_transpose_f16<<<tg, tb, 0, stream>>>(Wv, WvT);
  k_transpose_f16<<<tg, tb, 0, stream>>>(Wo, WoT);

  // 2) QKV projections (1/sqrt(H) folded into q)
  dim3 gg(64, 4);
  const float qscale = 0.044194173824159216f;  // 1/sqrt(512)
  k_gemm_wmma<<<gg, 256, 0, stream>>>(x16, WqT, bq, qscale, q16, nullptr, DH);
  k_gemm_wmma<<<gg, 256, 0, stream>>>(x16, WkT, bk, 1.0f,   k16, nullptr, DH);
  k_gemm_wmma<<<gg, 256, 0, stream>>>(x16, WvT, bv, 1.0f,   v16, nullptr, DH);

  // 3) flash attention (72 KB dynamic LDS)
  const size_t smem_bytes = (size_t)(2 * BN * DH + 8 * 16 * BN) * sizeof(_Float16);
  k_attention<<<64, 256, smem_bytes, stream>>>(q16, k16, v16, a16);

  // 4) output projection -> fp32 d_out
  k_gemm_wmma<<<gg, 256, 0, stream>>>(a16, WoT, bo, 1.0f, nullptr, out, DH);
}
